// GeoFieldMLP_11038065950940
// MI455X (gfx1250) — compile-verified, matrix operand order checked by reading, not run on hardware
//
#include <hip/hip_runtime.h>
#include <hip/hip_bf16.h>

// ---------------------------------------------------------------------------
// GeoFieldMLP fused forward for MI455X (gfx1250, wave32, WMMA bf16)
//   out = swiglu(x @ w_fusedT) @ (w0_down + lam * out_emb@((coords@R(q)T)@in_proj))T
// Sizes: x (16384,1024), w_fused (8192,1024), W (1024,4096), out (16384,1024)
// Strategy: bf16 WMMA (v_wmma_f32_16x16x32_bf16) for both GEMMs, f32 accum.
// Tiles staged with CDNA5 GLOBAL_LOAD_ASYNC_TO_LDS_B128 (ASYNCcnt) into a
// double-buffered LDS, no VGPR staging / ds_store in the pipeline.
// ---------------------------------------------------------------------------

typedef __attribute__((ext_vector_type(16))) __bf16 v16bf;
typedef __attribute__((ext_vector_type(8)))  float  v8f;

#define DIMV    1024
#define HIDDENV 4096
#define MROWS   16384      // B*S = 4*4096
#define NC      32

__device__ __forceinline__ unsigned short f32_to_bf16(float f) {
    union { float f; unsigned int u; } x; x.f = f;
    unsigned int u = x.u;
    u += 0x7fffu + ((u >> 16) & 1u);   // round-to-nearest-even
    return (unsigned short)(u >> 16);
}

union FragBF { v16bf v; uint4 q[2]; };

// Low 32 bits of a generic pointer into LDS == wave-relative LDS byte offset
// (flat LDS aperture keeps the offset in addr[31:0]; HW adds LDS_BASE).
__device__ __forceinline__ unsigned lds_off(const void* p) {
    return (unsigned)(unsigned long long)p;
}

// CDNA5 async global->LDS copy, 16B per lane, tracked by ASYNCcnt.
__device__ __forceinline__ void async_g2l_b128(unsigned lds, const void* g) {
    asm volatile("global_load_async_to_lds_b128 %0, %1, off"
                 :: "v"(lds), "v"(g) : "memory");
}

// ---------------------------------------------------------------- cast f32->bf16
__global__ void cast_f32_bf16(const float* __restrict__ in,
                              unsigned short* __restrict__ out, int n) {
    int i = blockIdx.x * blockDim.x + threadIdx.x;
    int stride = gridDim.x * blockDim.x;
    for (; i < n; i += stride) out[i] = f32_to_bf16(in[i]);
}

// ------------------------------------------------- P = (coords @ R(q)^T) @ in_proj
__global__ void geo_field_P(const float* __restrict__ coords,
                            const float* __restrict__ quat,
                            const float* __restrict__ in_proj,
                            float* __restrict__ P) {
    int h = blockIdx.x * blockDim.x + threadIdx.x;
    if (h >= HIDDENV) return;
    float qw = quat[0], qx = quat[1], qy = quat[2], qz = quat[3];
    float inv = rsqrtf(qw*qw + qx*qx + qy*qy + qz*qz);
    qw *= inv; qx *= inv; qy *= inv; qz *= inv;
    float R[3][3] = {
        {1.f - 2.f*(qy*qy + qz*qz), 2.f*(qx*qy - qw*qz),       2.f*(qx*qz + qw*qy)},
        {2.f*(qx*qy + qw*qz),       1.f - 2.f*(qx*qx + qz*qz), 2.f*(qy*qz - qw*qx)},
        {2.f*(qx*qz - qw*qy),       2.f*(qy*qz + qw*qx),       1.f - 2.f*(qx*qx + qy*qy)}};
    float ip0 = in_proj[0*HIDDENV + h];
    float ip1 = in_proj[1*HIDDENV + h];
    float ip2 = in_proj[2*HIDDENV + h];
    for (int c = 0; c < NC; ++c) {
        float c0 = coords[c*3+0], c1 = coords[c*3+1], c2 = coords[c*3+2];
        // rot_coords[c][j] = sum_k coords[c][k] * R[j][k]
        float r0 = c0*R[0][0] + c1*R[0][1] + c2*R[0][2];
        float r1 = c0*R[1][0] + c1*R[1][1] + c2*R[1][2];
        float r2 = c0*R[2][0] + c1*R[2][1] + c2*R[2][2];
        P[c*HIDDENV + h] = r0*ip0 + r1*ip1 + r2*ip2;
    }
}

// ------------------------------------- Wb = bf16(w0_down + lam * out_emb @ P)
__global__ void make_W_bf16(const float* __restrict__ w0,
                            const float* __restrict__ out_emb,
                            const float* __restrict__ P,
                            const float* __restrict__ lam_p,
                            unsigned short* __restrict__ Wb) {
    int idx = blockIdx.x * blockDim.x + threadIdx.x;
    if (idx >= DIMV * HIDDENV) return;
    int d = idx / HIDDENV, h = idx % HIDDENV;
    float lam = *lam_p;
    float acc = 0.f;
    for (int c = 0; c < NC; ++c)
        acc += out_emb[d*NC + c] * P[c*HIDDENV + h];
    Wb[idx] = f32_to_bf16(w0[idx] + lam * acc);
}

// ------------------------------------------------------------------- GEMM 1
// h = silu(x@Wg^T) * (x@Wd^T), block tile 128(M) x 64(N), K=1024, step 32.
// 8 waves: wm = wid&3 (32 rows each), wn = wid>>2 (32 cols each).
// LDS row stride 40 elems (80B): 16B-aligned rows, conflict-free b128 reads.
// Double-buffered LDS filled by async global->LDS B128 copies.
__global__ __launch_bounds__(256) void gemm1_swiglu(
        const unsigned short* __restrict__ xb,   // (MROWS, 1024) bf16
        const unsigned short* __restrict__ wfb,  // (8192, 1024) bf16
        unsigned short* __restrict__ hb)         // (MROWS, 4096) bf16
{
    // per buffer: A 128*40 (5120) | Bg 64*40 (2560) | Bd 64*40 (2560) ushorts
    __shared__ unsigned short smem[2 * 10240];

    const int tid  = threadIdx.x;
    const int lane = tid & 31, wid = tid >> 5;
    const int wm = wid & 3, wn = wid >> 2;
    const int m0 = blockIdx.y * 128;
    const int n0 = blockIdx.x * 64;
    const int K  = DIMV;

    const int lr = tid >> 2;        // 0..63 loader row
    const int lc = (tid & 3) * 8;   // loader elem col: 0,8,16,24

    unsigned dA0[2], dA1[2], dBg[2], dBd[2];
    for (int b = 0; b < 2; ++b) {
        unsigned short* base = smem + b * 10240;
        dA0[b] = lds_off(base +         lr        * 40 + lc);
        dA1[b] = lds_off(base +        (lr + 64)  * 40 + lc);
        dBg[b] = lds_off(base + 5120 +  lr        * 40 + lc);
        dBd[b] = lds_off(base + 7680 +  lr        * 40 + lc);
    }
    const unsigned short* gA0 = xb  + (size_t)(m0 + lr)           * K + lc;
    const unsigned short* gA1 = xb  + (size_t)(m0 + 64 + lr)      * K + lc;
    const unsigned short* gBg = wfb + (size_t)(n0 + lr)           * K + lc;
    const unsigned short* gBd = wfb + (size_t)(HIDDENV + n0 + lr) * K + lc;

    v8f accg[2][2], accd[2][2];
    for (int i = 0; i < 2; ++i)
        for (int j = 0; j < 2; ++j)
            for (int r = 0; r < 8; ++r) { accg[i][j][r] = 0.f; accd[i][j][r] = 0.f; }

    // preload tile 0 into buffer 0
    async_g2l_b128(dA0[0], gA0);
    async_g2l_b128(dA1[0], gA1);
    async_g2l_b128(dBg[0], gBg);
    async_g2l_b128(dBd[0], gBd);

    for (int k0 = 0, it = 0; k0 < K; k0 += 32, ++it) {
        const int cur = it & 1, nxt = cur ^ 1;
        __syncthreads();                 // all waves done reading buf[nxt]
        if (k0 + 32 < K) {
            const int kn = k0 + 32;
            async_g2l_b128(dA0[nxt], gA0 + kn);
            async_g2l_b128(dA1[nxt], gA1 + kn);
            async_g2l_b128(dBg[nxt], gBg + kn);
            async_g2l_b128(dBd[nxt], gBd + kn);
            asm volatile("s_wait_asynccnt 0x4" ::: "memory");  // cur tile landed
        } else {
            asm volatile("s_wait_asynccnt 0x0" ::: "memory");
        }
        __syncthreads();                 // cur tile visible to all waves

        const unsigned short* Al  = smem + cur * 10240;
        const unsigned short* Bgl = Al + 5120;
        const unsigned short* Bdl = Al + 7680;

        // Per-lane fragment layout (16-bit A, 16x32): lane<16 -> K 0-7 & 16-23,
        // lane>=16 -> K 8-15 & 24-31, row = lane&15.
        const int fr = lane & 15, fh = (lane >> 4) * 8;
        FragBF a[2], bg[2], bd[2];
        for (int i = 0; i < 2; ++i) {
            const unsigned short* p = &Al[(wm*32 + i*16 + fr) * 40 + fh];
            a[i].q[0] = *(const uint4*)(p);
            a[i].q[1] = *(const uint4*)(p + 16);
        }
        for (int j = 0; j < 2; ++j) {
            const unsigned short* pg = &Bgl[(wn*32 + j*16 + fr) * 40 + fh];
            bg[j].q[0] = *(const uint4*)(pg);
            bg[j].q[1] = *(const uint4*)(pg + 16);
            const unsigned short* pd = &Bdl[(wn*32 + j*16 + fr) * 40 + fh];
            bd[j].q[0] = *(const uint4*)(pd);
            bd[j].q[1] = *(const uint4*)(pd + 16);
        }
        for (int i = 0; i < 2; ++i)
            for (int j = 0; j < 2; ++j) {
                accg[i][j] = __builtin_amdgcn_wmma_f32_16x16x32_bf16(
                    false, a[i].v, false, bg[j].v, (short)0, accg[i][j], false, false);
                accd[i][j] = __builtin_amdgcn_wmma_f32_16x16x32_bf16(
                    false, a[i].v, false, bd[j].v, (short)0, accd[i][j], false, false);
            }
    }

    // Epilogue: fused SwiGLU, write h as bf16.
    // C layout: VGPR r, lanes 0-15 -> M=r, lanes 16-31 -> M=r+8, N=lane&15.
    const int fr = lane & 15, fm = (lane >> 4) * 8;
    for (int i = 0; i < 2; ++i)
        for (int j = 0; j < 2; ++j)
            for (int r = 0; r < 8; ++r) {
                float g  = accg[i][j][r];
                float dv = accd[i][j][r];
                float hv = (g / (1.f + __expf(-g))) * dv;
                int mrow = m0 + wm*32 + i*16 + fm + r;
                int ncol = n0 + wn*32 + j*16 + fr;
                hb[(size_t)mrow * HIDDENV + ncol] = f32_to_bf16(hv);
            }
}

// ------------------------------------------------------------------- GEMM 2
// out = h @ W^T, block tile 128x128, K=4096, step 32.
// 8 waves: wm = wid&3 (32 rows), wn = wid>>2 (64 cols -> 4 j-tiles).
__global__ __launch_bounds__(256) void gemm2_down(
        const unsigned short* __restrict__ hb,  // (MROWS, 4096) bf16
        const unsigned short* __restrict__ Wb,  // (1024, 4096) bf16
        float* __restrict__ out)                // (MROWS, 1024) f32
{
    // per buffer: A 128*40 (5120) | B 128*40 (5120) ushorts
    __shared__ unsigned short smem[2 * 10240];

    const int tid  = threadIdx.x;
    const int lane = tid & 31, wid = tid >> 5;
    const int wm = wid & 3, wn = wid >> 2;
    const int m0 = blockIdx.y * 128;
    const int n0 = blockIdx.x * 128;
    const int K  = HIDDENV;

    const int lr = tid >> 2;
    const int lc = (tid & 3) * 8;

    unsigned dA0[2], dA1[2], dB0[2], dB1[2];
    for (int b = 0; b < 2; ++b) {
        unsigned short* base = smem + b * 10240;
        dA0[b] = lds_off(base +         lr       * 40 + lc);
        dA1[b] = lds_off(base +        (lr + 64) * 40 + lc);
        dB0[b] = lds_off(base + 5120 +  lr       * 40 + lc);
        dB1[b] = lds_off(base + 5120 + (lr + 64) * 40 + lc);
    }
    const unsigned short* gA0 = hb + (size_t)(m0 + lr)      * K + lc;
    const unsigned short* gA1 = hb + (size_t)(m0 + 64 + lr) * K + lc;
    const unsigned short* gB0 = Wb + (size_t)(n0 + lr)      * K + lc;
    const unsigned short* gB1 = Wb + (size_t)(n0 + 64 + lr) * K + lc;

    v8f acc[2][4];
    for (int i = 0; i < 2; ++i)
        for (int j = 0; j < 4; ++j)
            for (int r = 0; r < 8; ++r) acc[i][j][r] = 0.f;

    async_g2l_b128(dA0[0], gA0);
    async_g2l_b128(dA1[0], gA1);
    async_g2l_b128(dB0[0], gB0);
    async_g2l_b128(dB1[0], gB1);

    for (int k0 = 0, it = 0; k0 < K; k0 += 32, ++it) {
        const int cur = it & 1, nxt = cur ^ 1;
        __syncthreads();
        if (k0 + 32 < K) {
            const int kn = k0 + 32;
            async_g2l_b128(dA0[nxt], gA0 + kn);
            async_g2l_b128(dA1[nxt], gA1 + kn);
            async_g2l_b128(dB0[nxt], gB0 + kn);
            async_g2l_b128(dB1[nxt], gB1 + kn);
            asm volatile("s_wait_asynccnt 0x4" ::: "memory");
        } else {
            asm volatile("s_wait_asynccnt 0x0" ::: "memory");
        }
        __syncthreads();

        const unsigned short* Al = smem + cur * 10240;
        const unsigned short* Bl = Al + 5120;

        const int fr = lane & 15, fh = (lane >> 4) * 8;
        FragBF a[2], b[4];
        for (int i = 0; i < 2; ++i) {
            const unsigned short* p = &Al[(wm*32 + i*16 + fr) * 40 + fh];
            a[i].q[0] = *(const uint4*)(p);
            a[i].q[1] = *(const uint4*)(p + 16);
        }
        for (int j = 0; j < 4; ++j) {
            const unsigned short* p = &Bl[(wn*64 + j*16 + fr) * 40 + fh];
            b[j].q[0] = *(const uint4*)(p);
            b[j].q[1] = *(const uint4*)(p + 16);
        }
        for (int i = 0; i < 2; ++i)
            for (int j = 0; j < 4; ++j)
                acc[i][j] = __builtin_amdgcn_wmma_f32_16x16x32_bf16(
                    false, a[i].v, false, b[j].v, (short)0, acc[i][j], false, false);
    }

    const int fr = lane & 15, fm = (lane >> 4) * 8;
    for (int i = 0; i < 2; ++i)
        for (int j = 0; j < 4; ++j)
            for (int r = 0; r < 8; ++r) {
                int mrow = m0 + wm*32 + i*16 + fm + r;
                int ncol = n0 + wn*64 + j*16 + fr;
                out[(size_t)mrow * DIMV + ncol] = acc[i][j][r];
            }
}

// ---------------------------------------------------------------------------
extern "C" void kernel_launch(void* const* d_in, const int* in_sizes, int n_in,
                              void* d_out, int out_size, void* d_ws, size_t ws_size,
                              hipStream_t stream) {
    const float* x       = (const float*)d_in[0];   // (4,4096,1024)
    const float* w_fused = (const float*)d_in[1];   // (8192,1024)
    const float* w0_down = (const float*)d_in[2];   // (1024,4096)
    const float* coords  = (const float*)d_in[3];   // (32,3)
    const float* quat    = (const float*)d_in[4];   // (4,)
    const float* in_proj = (const float*)d_in[5];   // (3,4096)
    const float* out_emb = (const float*)d_in[6];   // (1024,32)
    const float* lam     = (const float*)d_in[7];   // scalar
    float*       out     = (float*)d_out;           // (4,4096,1024)

    // Workspace carve-up (all offsets 256B-aligned): ~193.5 MB total
    char* ws = (char*)d_ws;
    unsigned short* xb  = (unsigned short*)ws; ws += (size_t)MROWS * DIMV   * 2; // 33.5 MB
    unsigned short* wfb = (unsigned short*)ws; ws += (size_t)2*HIDDENV*DIMV * 2; // 16.8 MB
    unsigned short* hb  = (unsigned short*)ws; ws += (size_t)MROWS * HIDDENV* 2; // 134 MB
    unsigned short* Wb  = (unsigned short*)ws; ws += (size_t)DIMV * HIDDENV * 2; //  8.4 MB
    float*          P   = (float*)ws;                                           //  0.5 MB

    // 1) bf16 casts of the two big GEMM inputs
    cast_f32_bf16<<<8192, 256, 0, stream>>>(x,       xb,  MROWS * DIMV);
    cast_f32_bf16<<<4096, 256, 0, stream>>>(w_fused, wfb, 2 * HIDDENV * DIMV);

    // 2) tiny geometric field + modulated down-projection weight (bf16)
    geo_field_P<<<HIDDENV / 256, 256, 0, stream>>>(coords, quat, in_proj, P);
    make_W_bf16<<<(DIMV * HIDDENV) / 256, 256, 0, stream>>>(w0_down, out_emb, P, lam, Wb);

    // 3) fused up-projection + SwiGLU  (M=16384, N=4096 pairs, K=1024)
    dim3 g1(HIDDENV / 64, MROWS / 128);
    gemm1_swiglu<<<g1, 256, 0, stream>>>(xb, wfb, hb);

    // 4) down-projection (M=16384, N=1024, K=4096)
    dim3 g2(DIMV / 128, MROWS / 128);
    gemm2_down<<<g2, 256, 0, stream>>>(hb, Wb, out);
}
